// PaiNNLayer_67027259621443
// MI455X (gfx1250) — matrix operand order repaired and checked by reading, not compiled, not run on hardware
//
#include <hip/hip_runtime.h>

#define N_NODES 50000
#define E_EDGES 800000
#define H       128
#define N_RBF   20
#define CUTOFF  5.0f
#define EPS     1e-9f
#define PI_F    3.14159265358979f

typedef __attribute__((ext_vector_type(16))) __bf16 v16bf;
typedef __attribute__((ext_vector_type(8)))  float  v8f;

union Frag {
    v16bf v;
    unsigned short s[16];
    unsigned u[8];
    uint4 q[2];
};

__device__ __forceinline__ unsigned short f2bf(float f) {
    union { float f; unsigned u; } x; x.f = f;
    unsigned r = x.u + 0x7FFFu + ((x.u >> 16) & 1u);   // RNE
    return (unsigned short)(r >> 16);
}
// pack two floats into a dword of two bf16 (RNE) with one v_perm_b32
__device__ __forceinline__ unsigned f2bf_pk(float lo, float hi) {
    union { float f; unsigned u; } a, b;
    a.f = lo; b.f = hi;
    unsigned ra = a.u + 0x7FFFu + ((a.u >> 16) & 1u);
    unsigned rb = b.u + 0x7FFFu + ((b.u >> 16) & 1u);
    return __builtin_amdgcn_perm(rb, ra, 0x07060302u); // {rb[31:16], ra[31:16]}
}
__device__ __forceinline__ float fastrcp(float x) {
    return __builtin_amdgcn_rcpf(x);
}
__device__ __forceinline__ float silu(float x) {
    return x * fastrcp(1.0f + __expf(-x));
}
__device__ __forceinline__ v8f splat8(float x) {
    v8f r;
    #pragma unroll
    for (int i = 0; i < 8; ++i) r[i] = x;
    return r;
}
// A-fragment K position for (pair p, half) in a 16x32 bf16 A tile
__device__ __forceinline__ int a_k0(int p, int half) {
    return (p < 4 ? 2 * p : 16 + 2 * (p - 4)) + 8 * half;
}
// packed B fragment: lane (0..31) == half*16+col; 16 bf16 contiguous
__device__ __forceinline__ void load_bfrag(const unsigned short* Bp, int kc, int n0,
                                           int lane, Frag& f) {
    const uint4* p = reinterpret_cast<const uint4*>(Bp + ((((kc * 8) + n0) * 32 + lane) << 4));
    f.q[0] = p[0];
    f.q[1] = p[1];
}
__device__ __forceinline__ v8f wmma_bf16(const Frag& a, const Frag& b, v8f c) {
    return __builtin_amdgcn_wmma_f32_16x16x32_bf16(false, a.v, false, b.v,
                                                   (short)0, c, false, false);
}

// ---------------- kernel 0: seed d_out (s-section = 0, v-section = v) -------
__global__ __launch_bounds__(256) void k_init(float* out, const float* vin,
                                              int nh, int total) {
    int idx = blockIdx.x * 256 + threadIdx.x;
    if (idx >= total) return;
    out[idx] = (idx < nh) ? 0.0f : vin[idx - nh];
}

// ---------------- kernel 1: pack fp32 [K,128] weight into bf16 B-fragment layout
__global__ __launch_bounds__(256) void k_pack(const float* W, unsigned short* out,
                                              int K, int total) {
    int t = blockIdx.x * 256 + threadIdx.x;
    if (t >= total) return;
    int q    = t & 15;
    int rest = t >> 4;
    int lhc  = rest & 31;          // half*16 + col
    int n0   = (rest >> 5) & 7;
    int kc   = rest >> 8;
    int half = lhc >> 4, col = lhc & 15;
    int k = kc * 32 + (q < 8 ? half * 8 + q : 16 + half * 8 + (q - 8));
    int n = n0 * 16 + col;
    float val = (k < K) ? W[k * H + n] : 0.0f;
    out[t] = f2bf(val);
}

// ---------------- kernel 2: per-edge pipeline, 16 edges per wave -----------
__global__ __launch_bounds__(128) void k_edge(
    const float* __restrict__ s, const float* __restrict__ vin,
    const float* __restrict__ pos, const int* __restrict__ ei,
    const float* __restrict__ phi_b1, const float* __restrict__ phi_b2,
    const float* __restrict__ Ws_b,  const float* __restrict__ Wv_b,
    const unsigned short* __restrict__ phiW1p, const unsigned short* __restrict__ phiW2p,
    const unsigned short* __restrict__ WsP,    const unsigned short* __restrict__ WvP,
    float* ms_out, float* mv_out) {

    __shared__ unsigned short Ab[4][16 * 32];    // rbf staging (bf16)
    __shared__ unsigned short Hb[4][16 * 128];   // silu(phi1) staging (bf16)
    __shared__ float          Wb[4][16 * 128];   // w_ij staging (fp32)
    __shared__ float unitS[4][16 * 3];
    __shared__ int   idxiS[4][16];
    __shared__ int   idxjS[4][16];

    const int wave = threadIdx.x >> 5;
    const int lane = threadIdx.x & 31;
    const int row  = lane & 15;
    const int half = lane >> 4;
    const int tile = blockIdx.x * 4 + wave;
    const int e0   = tile * 16;

    // ---- phase 1: geometry + RBF (lanes 0..15, one edge each) ----
    if (lane < 16) {
        int e = e0 + lane;
        int ii = ei[e];
        int jj = ei[E_EDGES + e];
        float dx = pos[jj * 3 + 0] - pos[ii * 3 + 0];
        float dy = pos[jj * 3 + 1] - pos[ii * 3 + 1];
        float dz = pos[jj * 3 + 2] - pos[ii * 3 + 2];
        float d  = sqrtf(dx * dx + dy * dy + dz * dz);
        float inv = fastrcp(d + EPS);
        unitS[wave][lane * 3 + 0] = dx * inv;
        unitS[wave][lane * 3 + 1] = dy * inv;
        unitS[wave][lane * 3 + 2] = dz * inv;
        idxiS[wave][lane] = ii;
        idxjS[wave][lane] = jj;
        float cut = (d < CUTOFF) ? 0.5f * (__cosf(PI_F * d / CUTOFF) + 1.0f) : 0.0f;
        float sc  = (d > 0.0f) ? cut * fastrcp(d) : 0.0f;
        #pragma unroll
        for (int n = 1; n <= N_RBF; ++n)
            Ab[wave][lane * 32 + n - 1] = f2bf(__sinf(d * (float)n * (PI_F / CUTOFF)) * sc);
        #pragma unroll
        for (int k = N_RBF; k < 32; ++k) Ab[wave][lane * 32 + k] = 0;
    }
    __syncthreads();

    const int coln = lane & 15;

    // ---- phase 2: w = silu(rbf @ phi_w1 + b1) -> Hb (bias rides in C) ----
    {
        Frag a;
        #pragma unroll
        for (int p = 0; p < 8; ++p)
            a.u[p] = *reinterpret_cast<const unsigned*>(&Ab[wave][row * 32 + a_k0(p, half)]);
        #pragma unroll
        for (int n0 = 0; n0 < 8; ++n0) {
            int n = n0 * 16 + coln;
            Frag b; load_bfrag(phiW1p, 0, n0, lane, b);
            v8f acc = wmma_bf16(a, b, splat8(phi_b1[n]));
            #pragma unroll
            for (int r = 0; r < 8; ++r) {
                int m = half * 8 + r;
                Hb[wave][m * 128 + n] = f2bf(silu(acc[r]));
            }
        }
    }
    __syncthreads();

    // ---- phase 3: w_ij = h @ phi_w2 + b2 -> Wb ----
    {
        v8f acc[8];
        #pragma unroll
        for (int n0 = 0; n0 < 8; ++n0) acc[n0] = splat8(phi_b2[n0 * 16 + coln]);
        #pragma unroll
        for (int kc = 0; kc < 4; ++kc) {
            Frag a;
            #pragma unroll
            for (int p = 0; p < 8; ++p)
                a.u[p] = *reinterpret_cast<const unsigned*>(
                    &Hb[wave][row * 128 + kc * 32 + a_k0(p, half)]);
            #pragma unroll
            for (int n0 = 0; n0 < 8; ++n0) {
                Frag b; load_bfrag(phiW2p, kc, n0, lane, b);
                acc[n0] = wmma_bf16(a, b, acc[n0]);
            }
        }
        #pragma unroll
        for (int n0 = 0; n0 < 8; ++n0) {
            int n = n0 * 16 + coln;
            #pragma unroll
            for (int r = 0; r < 8; ++r) {
                int m = half * 8 + r;
                Wb[wave][m * 128 + n] = acc[n0][r];
            }
        }
    }
    __syncthreads();

    const int jrow = idxjS[wave][row];

    // ---- phase 4: m_s = (s_j @ Ws + b) * w_ij, scatter-add ----
    {
        const float* srow = s + (long)jrow * H;
        v8f acc[8];
        #pragma unroll
        for (int n0 = 0; n0 < 8; ++n0) acc[n0] = splat8(Ws_b[n0 * 16 + coln]);
        #pragma unroll
        for (int kc = 0; kc < 4; ++kc) {
            Frag a;
            #pragma unroll
            for (int p = 0; p < 8; ++p) {
                int k = kc * 32 + a_k0(p, half);
                float2 t = *reinterpret_cast<const float2*>(srow + k);
                a.u[p] = f2bf_pk(t.x, t.y);
            }
            #pragma unroll
            for (int n0 = 0; n0 < 8; ++n0) {
                Frag b; load_bfrag(WsP, kc, n0, lane, b);
                acc[n0] = wmma_bf16(a, b, acc[n0]);
            }
        }
        #pragma unroll
        for (int n0 = 0; n0 < 8; ++n0) {
            int n = n0 * 16 + coln;
            #pragma unroll
            for (int r = 0; r < 8; ++r) {
                int m = half * 8 + r;
                float val = acc[n0][r] * Wb[wave][m * 128 + n];
                atomicAdd(&ms_out[(long)idxiS[wave][m] * H + n], val);
            }
        }
    }

    // ---- phase 5: m_v = (proj @ Wv + b) * w_ij; scatter unit x m_v ----
    {
        const float* vrow = vin + (long)jrow * 3 * H;
        float u0 = unitS[wave][row * 3 + 0];
        float u1 = unitS[wave][row * 3 + 1];
        float u2 = unitS[wave][row * 3 + 2];
        v8f acc[8];
        #pragma unroll
        for (int n0 = 0; n0 < 8; ++n0) acc[n0] = splat8(Wv_b[n0 * 16 + coln]);
        #pragma unroll
        for (int kc = 0; kc < 4; ++kc) {
            Frag a;
            #pragma unroll
            for (int p = 0; p < 8; ++p) {
                int k = kc * 32 + a_k0(p, half);
                float2 t0 = *reinterpret_cast<const float2*>(vrow + 0 * H + k);
                float2 t1 = *reinterpret_cast<const float2*>(vrow + 1 * H + k);
                float2 t2 = *reinterpret_cast<const float2*>(vrow + 2 * H + k);
                a.u[p] = f2bf_pk(u0 * t0.x + u1 * t1.x + u2 * t2.x,
                                 u0 * t0.y + u1 * t1.y + u2 * t2.y);
            }
            #pragma unroll
            for (int n0 = 0; n0 < 8; ++n0) {
                Frag b; load_bfrag(WvP, kc, n0, lane, b);
                acc[n0] = wmma_bf16(a, b, acc[n0]);
            }
        }
        #pragma unroll
        for (int n0 = 0; n0 < 8; ++n0) {
            int n = n0 * 16 + coln;
            #pragma unroll
            for (int r = 0; r < 8; ++r) {
                int m = half * 8 + r;
                float mv = acc[n0][r] * Wb[wave][m * 128 + n];
                int inode = idxiS[wave][m];
                float w0 = unitS[wave][m * 3 + 0];
                float w1 = unitS[wave][m * 3 + 1];
                float w2 = unitS[wave][m * 3 + 2];
                float* base = mv_out + (long)inode * 3 * H + n;
                atomicAdd(base + 0 * H, w0 * mv);
                atomicAdd(base + 1 * H, w1 * mv);
                atomicAdd(base + 2 * H, w2 * mv);
            }
        }
    }
}

// ---------------- kernel 3: node update s_out = s + MLP(m_s) ---------------
__global__ __launch_bounds__(128) void k_node(
    float* ms, const float* __restrict__ s_in,
    const unsigned short* __restrict__ Us1P, const unsigned short* __restrict__ Us2P,
    const float* __restrict__ Us_b1, const float* __restrict__ Us_b2) {

    __shared__ unsigned short Hb[4][16 * 128];

    const int wave = threadIdx.x >> 5;
    const int lane = threadIdx.x & 31;
    const int row  = lane & 15;
    const int half = lane >> 4;
    const int NT   = N_NODES / 16;
    int tile = blockIdx.x * 4 + wave;
    bool active = tile < NT;
    int t = active ? tile : 0;
    const int r0 = t * 16;
    const int coln = lane & 15;

    // GEMM1: silu(m_s @ Us_w1 + b1) -> Hb
    {
        const float* arow = ms + (long)(r0 + row) * H;
        v8f acc[8];
        #pragma unroll
        for (int n0 = 0; n0 < 8; ++n0) acc[n0] = splat8(Us_b1[n0 * 16 + coln]);
        #pragma unroll
        for (int kc = 0; kc < 4; ++kc) {
            Frag a;
            #pragma unroll
            for (int p = 0; p < 8; ++p) {
                int k = kc * 32 + a_k0(p, half);
                float2 tv = *reinterpret_cast<const float2*>(arow + k);
                a.u[p] = f2bf_pk(tv.x, tv.y);
            }
            #pragma unroll
            for (int n0 = 0; n0 < 8; ++n0) {
                Frag b; load_bfrag(Us1P, kc, n0, lane, b);
                acc[n0] = wmma_bf16(a, b, acc[n0]);
            }
        }
        #pragma unroll
        for (int n0 = 0; n0 < 8; ++n0) {
            int n = n0 * 16 + coln;
            #pragma unroll
            for (int r = 0; r < 8; ++r) {
                int m = half * 8 + r;
                Hb[wave][m * 128 + n] = f2bf(silu(acc[n0][r]));
            }
        }
    }
    __syncthreads();

    // GEMM2: s_out = s + (h @ Us_w2 + b2), overwrite ms rows
    {
        v8f acc[8];
        #pragma unroll
        for (int n0 = 0; n0 < 8; ++n0) acc[n0] = splat8(Us_b2[n0 * 16 + coln]);
        #pragma unroll
        for (int kc = 0; kc < 4; ++kc) {
            Frag a;
            #pragma unroll
            for (int p = 0; p < 8; ++p)
                a.u[p] = *reinterpret_cast<const unsigned*>(
                    &Hb[wave][row * 128 + kc * 32 + a_k0(p, half)]);
            #pragma unroll
            for (int n0 = 0; n0 < 8; ++n0) {
                Frag b; load_bfrag(Us2P, kc, n0, lane, b);
                acc[n0] = wmma_bf16(a, b, acc[n0]);
            }
        }
        if (active) {
            #pragma unroll
            for (int n0 = 0; n0 < 8; ++n0) {
                int n = n0 * 16 + coln;
                #pragma unroll
                for (int r = 0; r < 8; ++r) {
                    int m = half * 8 + r;
                    long node = r0 + m;
                    ms[node * H + n] = s_in[node * H + n] + acc[n0][r];
                }
            }
        }
    }
}

extern "C" void kernel_launch(void* const* d_in, const int* in_sizes, int n_in,
                              void* d_out, int out_size, void* d_ws, size_t ws_size,
                              hipStream_t stream) {
    const float* s      = (const float*)d_in[0];
    const float* v      = (const float*)d_in[1];
    const float* pos    = (const float*)d_in[2];
    const int*   ei     = (const int*)d_in[3];
    const float* phi_w1 = (const float*)d_in[4];
    const float* phi_b1 = (const float*)d_in[5];
    const float* phi_w2 = (const float*)d_in[6];
    const float* phi_b2 = (const float*)d_in[7];
    const float* Ws_w   = (const float*)d_in[8];
    const float* Ws_b   = (const float*)d_in[9];
    const float* Wv_w   = (const float*)d_in[10];
    const float* Wv_b   = (const float*)d_in[11];
    const float* Us_w1  = (const float*)d_in[12];
    const float* Us_b1  = (const float*)d_in[13];
    const float* Us_w2  = (const float*)d_in[14];
    const float* Us_b2  = (const float*)d_in[15];

    float* out  = (float*)d_out;
    float* ms   = out;                       // [N,H] — accumulates m_s, becomes s_out
    float* mv   = out + (long)N_NODES * H;   // [N,3,H] — seeded with v, accumulates m_v

    unsigned short* wsp = (unsigned short*)d_ws;
    unsigned short* phiW1p = wsp;                 // 32*128  = 4096
    unsigned short* phiW2p = phiW1p + 4096;       // 128*128 = 16384
    unsigned short* WsP    = phiW2p + 16384;
    unsigned short* WvP    = WsP    + 16384;
    unsigned short* Us1P   = WvP    + 16384;
    unsigned short* Us2P   = Us1P   + 16384;

    // seed output
    {
        int nh = N_NODES * H;
        int total = nh + N_NODES * 3 * H;
        k_init<<<(total + 255) / 256, 256, 0, stream>>>(out, v, nh, total);
    }
    // pack weights to bf16 B-fragment layout (stays resident in L2)
    k_pack<<<(4096  + 255) / 256, 256, 0, stream>>>(phi_w1, phiW1p, N_RBF, 4096);
    k_pack<<<(16384 + 255) / 256, 256, 0, stream>>>(phi_w2, phiW2p, H, 16384);
    k_pack<<<(16384 + 255) / 256, 256, 0, stream>>>(Ws_w,   WsP,    H, 16384);
    k_pack<<<(16384 + 255) / 256, 256, 0, stream>>>(Wv_w,   WvP,    H, 16384);
    k_pack<<<(16384 + 255) / 256, 256, 0, stream>>>(Us_w1,  Us1P,   H, 16384);
    k_pack<<<(16384 + 255) / 256, 256, 0, stream>>>(Us_w2,  Us2P,   H, 16384);

    // edge kernel: 50000 tiles of 16 edges, 4 waves/block
    k_edge<<<E_EDGES / 16 / 4, 128, 0, stream>>>(
        s, v, pos, ei, phi_b1, phi_b2, Ws_b, Wv_b,
        phiW1p, phiW2p, WsP, WvP, ms, mv);

    // node kernel: 3125 tiles of 16 nodes
    int nt = N_NODES / 16;
    k_node<<<(nt + 3) / 4, 128, 0, stream>>>(ms, s, Us1P, Us2P, Us_b1, Us_b2);
}
